// ExampleEncoderLayer_68401649156347
// MI455X (gfx1250) — compile-verified, matrix operand order checked by reading, not run on hardware
//
#include <hip/hip_runtime.h>
#include <math.h>

typedef __attribute__((ext_vector_type(16))) __bf16 v16bf;
typedef __attribute__((ext_vector_type(8)))  __bf16 v8bf;
typedef __attribute__((ext_vector_type(8)))  float  v8f;

#define EPSV 1e-5f

__device__ __forceinline__ __bf16 f2bf(float f) {
  unsigned u = __builtin_bit_cast(unsigned, f);
  unsigned r = (u + 0x7FFFu + ((u >> 16) & 1u)) >> 16;
  unsigned short s = (unsigned short)r;
  return __builtin_bit_cast(__bf16, s);
}

// ---------------------------------------------------------------------------
// Tiled wave32 WMMA GEMM:  C[M,N] (+)= alpha * A[M,K] x B
//   BT=true : B stored row-major (N,K)  ->  C = A * B^T   (torch Linear / 1x1 conv)
//   BT=false: B stored row-major (K,N)  ->  C = A * B
// Strided-batched via (sA,sB,sC); grid.z = batch.
// Block tile 128x128, BK=32, 8 waves (2x4), each wave 64x32 = 4x2 WMMA frags.
// ---------------------------------------------------------------------------
template<bool BT, bool ACCUM>
__global__ __launch_bounds__(256)
void wmma_gemm(const __bf16* __restrict__ A, int lda, long long sA,
               const __bf16* __restrict__ B, int ldb, long long sB,
               float* __restrict__ C, int ldc, long long sC,
               int M, int Nn, int K, float alpha)
{
  constexpr int BM = 128, BN = 128, BK = 32;
  constexpr int LDA_S = BK + 8;                       // 40 halves (80B rows, 16B aligned)
  constexpr int BSZ   = BT ? (BM * LDA_S) : (BK * (BN + 8));
  __shared__ __align__(16) __bf16 Asm[BM * LDA_S];
  __shared__ __align__(16) __bf16 Bsm[BSZ];

  const int tid  = threadIdx.x;
  const int lane = tid & 31;
  const int wid  = tid >> 5;       // 0..7
  const int wm   = wid >> 2;       // 0..1
  const int wn   = wid & 3;        // 0..3
  const int half = lane >> 4;      // 0..1
  const int lrow = lane & 15;      // 0..15
  const int tileM = blockIdx.y * BM;
  const int tileN = blockIdx.x * BN;
  const long long bz = blockIdx.z;

  const __bf16* Ab = A + bz * sA;
  const __bf16* Bb = B + bz * sB;
  float*        Cb = C + bz * sC;

  v8f acc[4][2];
#pragma unroll
  for (int mi = 0; mi < 4; ++mi)
#pragma unroll
    for (int ni = 0; ni < 2; ++ni) {
      v8f z = {0.f, 0.f, 0.f, 0.f, 0.f, 0.f, 0.f, 0.f};
      acc[mi][ni] = z;
    }

  for (int k0 = 0; k0 < K; k0 += BK) {
    // ---- A tile: 128 rows x 32 halves; 2 threads/row, 16 halves each ----
    {
      int r   = tid >> 1;
      int cbk = (tid & 1) * 16;
      int grow = tileM + r;
      uint4 d0 = {0, 0, 0, 0}, d1 = {0, 0, 0, 0};
      if (grow < M) {
        const __bf16* src = Ab + (long long)grow * lda + k0 + cbk;
        d0 = *(const uint4*)(src);
        d1 = *(const uint4*)(src + 8);
        if (k0 + BK < K) __builtin_prefetch(src + BK, 0, 1);   // global_prefetch_b8
      }
      *(uint4*)&Asm[r * LDA_S + cbk]     = d0;
      *(uint4*)&Asm[r * LDA_S + cbk + 8] = d1;
    }
    // ---- B tile ----
    if constexpr (BT) {
      int r   = tid >> 1;
      int cbk = (tid & 1) * 16;
      int gn  = tileN + r;
      uint4 d0 = {0, 0, 0, 0}, d1 = {0, 0, 0, 0};
      if (gn < Nn) {
        const __bf16* src = Bb + (long long)gn * ldb + k0 + cbk;
        d0 = *(const uint4*)(src);
        d1 = *(const uint4*)(src + 8);
        if (k0 + BK < K) __builtin_prefetch(src + BK, 0, 1);
      }
      *(uint4*)&Bsm[r * LDA_S + cbk]     = d0;
      *(uint4*)&Bsm[r * LDA_S + cbk + 8] = d1;
    } else {
      int kr = tid >> 3;             // 0..31
      int nb = (tid & 7) * 16;       // 0..112
      int gn = tileN + nb;
      uint4 d0 = {0, 0, 0, 0}, d1 = {0, 0, 0, 0};
      if (gn < Nn) {
        const __bf16* src = Bb + (long long)(k0 + kr) * ldb + gn;
        d0 = *(const uint4*)(src);
        d1 = *(const uint4*)(src + 8);
      }
      *(uint4*)&Bsm[kr * (BN + 8) + nb]     = d0;
      *(uint4*)&Bsm[kr * (BN + 8) + nb + 8] = d1;
    }
    __syncthreads();

    // ---- fragment loads (ISA 16-bit A layout: lane=M, half selects K groups) ----
    v16bf af[4];
#pragma unroll
    for (int mi = 0; mi < 4; ++mi) {
      const __bf16* p = &Asm[(wm * 64 + mi * 16 + lrow) * LDA_S + half * 8];
      v8bf lo = *(const v8bf*)(p);
      v8bf hi = *(const v8bf*)(p + 16);
      v16bf t;
#pragma unroll
      for (int e = 0; e < 8; ++e) { t[e] = lo[e]; t[e + 8] = hi[e]; }
      af[mi] = t;
    }
    v16bf bfr[2];
#pragma unroll
    for (int ni = 0; ni < 2; ++ni) {
      if constexpr (BT) {
        const __bf16* p = &Bsm[(wn * 32 + ni * 16 + lrow) * LDA_S + half * 8];
        v8bf lo = *(const v8bf*)(p);
        v8bf hi = *(const v8bf*)(p + 16);
        v16bf t;
#pragma unroll
        for (int e = 0; e < 8; ++e) { t[e] = lo[e]; t[e + 8] = hi[e]; }
        bfr[ni] = t;
      } else {
        int n = wn * 32 + ni * 16 + lrow;
        v16bf t;
#pragma unroll
        for (int e = 0; e < 16; ++e) {
          int kk = ((e >> 3) << 4) + half * 8 + (e & 7);
          t[e] = Bsm[kk * (BN + 8) + n];
        }
        bfr[ni] = t;
      }
    }
#pragma unroll
    for (int mi = 0; mi < 4; ++mi)
#pragma unroll
      for (int ni = 0; ni < 2; ++ni)
        acc[mi][ni] = __builtin_amdgcn_wmma_f32_16x16x32_bf16(
            false, af[mi], false, bfr[ni], (short)0, acc[mi][ni], false, false);
    __syncthreads();
  }

  // ---- store (C layout: VGPR r -> M = base + r + 8*half; lane%16 = N) ----
#pragma unroll
  for (int mi = 0; mi < 4; ++mi) {
    int rowbase = tileM + wm * 64 + mi * 16 + half * 8;
#pragma unroll
    for (int ni = 0; ni < 2; ++ni) {
      int col = tileN + wn * 32 + ni * 16 + lrow;
      if (col < Nn) {
#pragma unroll
        for (int r = 0; r < 8; ++r) {
          int row = rowbase + r;
          if (row < M) {
            long long idx = (long long)row * ldc + col;
            float val = alpha * acc[mi][ni][r];
            if (ACCUM) val += Cb[idx];
            Cb[idx] = val;
          }
        }
      }
    }
  }
}

// ---------------------------------------------------------------------------
// BN (running stats) + (N,C,L) -> (N,L,C) transpose, f32 + bf16 outputs
// ---------------------------------------------------------------------------
__global__ void bn_transpose(const float* __restrict__ x,
                             const float* __restrict__ g, const float* __restrict__ b,
                             const float* __restrict__ m, const float* __restrict__ v,
                             float* __restrict__ hf, __bf16* __restrict__ hb,
                             int Cc, int Ll)
{
  __shared__ float tile[32][33];
  int n = blockIdx.z;
  int c0 = blockIdx.y * 32, l0 = blockIdx.x * 32;
  int tx = threadIdx.x, ty = threadIdx.y;
  for (int i = ty; i < 32; i += 8) {
    int c = c0 + i;
    float sc = g[c] * rsqrtf(v[c] + EPSV);
    float bs = b[c] - m[c] * sc;
    tile[i][tx] = x[((long long)(n * Cc + c)) * Ll + l0 + tx] * sc + bs;
  }
  __syncthreads();
  for (int i = ty; i < 32; i += 8) {
    int l = l0 + i, c = c0 + tx;
    float val = tile[tx][i];
    long long oi = ((long long)(n * Ll + l)) * Cc + c;
    hf[oi] = val;
    hb[oi] = f2bf(val);
  }
}

__global__ void cvt_bf16_strided(const float* __restrict__ in, __bf16* __restrict__ out,
                                 long long n, int stride, int off)
{
  long long i = (long long)blockIdx.x * 256 + threadIdx.x;
  if (i < n) out[i] = f2bf(in[i * stride + off]);
}

// (N,L,H,D) f32 -> (N,H,L,D) bf16
__global__ void pack_heads(const float* __restrict__ in, __bf16* __restrict__ out,
                           int NB, int Ll, int Hh, int Dd)
{
  long long total = (long long)NB * Ll * Hh * Dd;
  long long i = (long long)blockIdx.x * 256 + threadIdx.x;
  if (i >= total) return;
  int d  = (int)(i % Dd);
  int l  = (int)((i / Dd) % Ll);
  int hh = (int)((i / ((long long)Dd * Ll)) % Hh);
  int n  = (int)(i / ((long long)Dd * Ll * Hh));
  out[i] = f2bf(in[((long long)(n * Ll + l) * Hh + hh) * Dd + d]);
}

// (N,H,L,D) f32 -> (N,L,C) bf16
__global__ void unpack_heads(const float* __restrict__ in, __bf16* __restrict__ out,
                             int NB, int Ll, int Hh, int Dd)
{
  int Cc = Hh * Dd;
  long long total = (long long)NB * Ll * Cc;
  long long i = (long long)blockIdx.x * 256 + threadIdx.x;
  if (i >= total) return;
  int c = (int)(i % Cc);
  int l = (int)((i / Cc) % Ll);
  int n = (int)(i / ((long long)Cc * Ll));
  int hh = c / Dd, d = c % Dd;
  out[i] = f2bf(in[((long long)(n * Hh + hh) * Ll + l) * Dd + d]);
}

// Row softmax f32 -> bf16 (one block per row)
__global__ __launch_bounds__(256)
void softmax_bf16(const float* __restrict__ in, __bf16* __restrict__ out, int Lk)
{
  long long row = blockIdx.x;
  const float* p = in + row * Lk;
  __bf16* o = out + row * Lk;
  __shared__ float red[8];
  int tid = threadIdx.x, lane = tid & 31, wid = tid >> 5;
  float mx = -3.4e38f;
  for (int i = tid; i < Lk; i += 256) mx = fmaxf(mx, p[i]);
  for (int off = 16; off > 0; off >>= 1) mx = fmaxf(mx, __shfl_xor(mx, off, 32));
  if (lane == 0) red[wid] = mx;
  __syncthreads();
  if (wid == 0) {
    float t = (lane < 8) ? red[lane] : -3.4e38f;
    for (int off = 4; off > 0; off >>= 1) t = fmaxf(t, __shfl_xor(t, off, 32));
    if (lane == 0) red[0] = t;
  }
  __syncthreads();
  mx = red[0];
  float s = 0.f;
  for (int i = tid; i < Lk; i += 256) s += __expf(p[i] - mx);
  for (int off = 16; off > 0; off >>= 1) s += __shfl_xor(s, off, 32);
  __syncthreads();
  if (lane == 0) red[wid] = s;
  __syncthreads();
  if (wid == 0) {
    float t = (lane < 8) ? red[lane] : 0.f;
    for (int off = 4; off > 0; off >>= 1) t += __shfl_xor(t, off, 32);
    if (lane == 0) red[0] = t;
  }
  __syncthreads();
  float inv = 1.0f / red[0];
  for (int i = tid; i < Lk; i += 256) o[i] = f2bf(__expf(p[i] - mx) * inv);
}

__global__ void residual_add(const float* __restrict__ a, const float* __restrict__ b,
                             float* __restrict__ hf, __bf16* __restrict__ hb, long long n)
{
  long long i = (long long)blockIdx.x * 256 + threadIdx.x;
  if (i >= n) return;
  float v = a[i] + b[i];
  hf[i] = v;
  hb[i] = f2bf(v);
}

// BN (+folded conv bias) + optional ReLU; optional zero-padded sequence rows for k=3 conv
__global__ void bn_act_bf16(const float* __restrict__ y,
                            const float* __restrict__ g, const float* __restrict__ b,
                            const float* __restrict__ m, const float* __restrict__ v,
                            const float* __restrict__ cb,
                            __bf16* __restrict__ out, int NB, int Ll, int Cc, int pad, int relu)
{
  int Lp = Ll + 2 * pad;
  long long total = (long long)NB * Lp * Cc;
  long long i = (long long)blockIdx.x * 256 + threadIdx.x;
  if (i >= total) return;
  int c  = (int)(i % Cc);
  int rr = (int)((i / Cc) % Lp);
  int n  = (int)(i / ((long long)Cc * Lp));
  if (pad && (rr == 0 || rr == Ll + 1)) { out[i] = f2bf(0.f); return; }
  int l = rr - pad;
  float sc = g[c] * rsqrtf(v[c] + EPSV);
  float bs = b[c] + sc * (cb[c] - m[c]);
  float val = sc * y[((long long)(n * Ll + l)) * Cc + c] + bs;
  if (relu) val = fmaxf(val, 0.f);
  out[i] = f2bf(val);
}

// BN3 + residual + instance-norm(L) + ReLU + maxpool2 -> (N,C,L/2)
__global__ __launch_bounds__(256)
void final_fuse(const float* __restrict__ y3, const float* __restrict__ hres,
                const float* __restrict__ g, const float* __restrict__ b,
                const float* __restrict__ m, const float* __restrict__ v,
                const float* __restrict__ cb,
                float* __restrict__ out, int Cc, int Ll)
{
  __shared__ float zbuf[1024];
  __shared__ float red[16];
  int n = blockIdx.x / Cc, c = blockIdx.x % Cc;
  int tid = threadIdx.x, lane = tid & 31, wid = tid >> 5;
  float sc = g[c] * rsqrtf(v[c] + EPSV);
  float bs = b[c] + sc * (cb[c] - m[c]);
  float s = 0.f, s2 = 0.f;
  for (int l = tid; l < Ll; l += 256) {
    long long idx = ((long long)(n * Ll + l)) * Cc + c;
    float z = sc * y3[idx] + bs + hres[idx];
    zbuf[l] = z; s += z; s2 += z * z;
  }
  for (int off = 16; off > 0; off >>= 1) { s += __shfl_xor(s, off, 32); s2 += __shfl_xor(s2, off, 32); }
  if (lane == 0) { red[wid] = s; red[wid + 8] = s2; }
  __syncthreads();
  if (wid == 0) {
    float a  = (lane < 8) ? red[lane] : 0.f;
    float a2 = (lane < 8) ? red[lane + 8] : 0.f;
    for (int off = 4; off > 0; off >>= 1) { a += __shfl_xor(a, off, 32); a2 += __shfl_xor(a2, off, 32); }
    if (lane == 0) { red[0] = a; red[1] = a2; }
  }
  __syncthreads();
  float mean = red[0] / Ll;
  float var  = red[1] / Ll - mean * mean;
  float rs   = rsqrtf(var + EPSV);
  for (int l2 = tid; l2 < Ll / 2; l2 += 256) {
    float a = fmaxf((zbuf[2 * l2]     - mean) * rs, 0.f);
    float d = fmaxf((zbuf[2 * l2 + 1] - mean) * rs, 0.f);
    out[((long long)(n * Cc + c)) * (Ll / 2) + l2] = fmaxf(a, d);
  }
}

// ---------------------------------------------------------------------------
extern "C" void kernel_launch(void* const* d_in, const int* in_sizes, int n_in,
                              void* d_out, int out_size, void* d_ws, size_t ws_size,
                              hipStream_t stream)
{
  constexpr int NB = 4, Cc = 1024, Ll = 1024, MD = 1024, NH = 16, DH = 64;
  constexpr long long NLC = (long long)NB * Ll * Cc;           // 4,194,304
  constexpr long long ATT = (long long)NB * NH * Ll * Ll;      // 67,108,864

  const float* x    = (const float*)d_in[0];
  const float* ng   = (const float*)d_in[1];
  const float* nb_  = (const float*)d_in[2];
  const float* nm   = (const float*)d_in[3];
  const float* nv   = (const float*)d_in[4];
  const float* wq   = (const float*)d_in[5];
  const float* wk   = (const float*)d_in[6];
  const float* wv_  = (const float*)d_in[7];
  const float* wo   = (const float*)d_in[8];
  const float* cw1  = (const float*)d_in[9];
  const float* cb1  = (const float*)d_in[10];
  const float* cw2  = (const float*)d_in[11];
  const float* cb2  = (const float*)d_in[12];
  const float* cw3  = (const float*)d_in[13];
  const float* cb3  = (const float*)d_in[14];
  const float* bn1g = (const float*)d_in[15];
  const float* bn1b = (const float*)d_in[16];
  const float* bn1m = (const float*)d_in[17];
  const float* bn1v = (const float*)d_in[18];
  const float* bn2g = (const float*)d_in[19];
  const float* bn2b = (const float*)d_in[20];
  const float* bn2m = (const float*)d_in[21];
  const float* bn2v = (const float*)d_in[22];
  const float* bn3g = (const float*)d_in[23];
  const float* bn3b = (const float*)d_in[24];
  const float* bn3m = (const float*)d_in[25];
  const float* bn3v = (const float*)d_in[26];

  char* wsb = (char*)d_ws;
  size_t off = 0;
  auto alloc = [&](size_t bytes) -> void* {
    void* p = wsb + off;
    off = (off + bytes + 255) & ~(size_t)255;
    return p;
  };

  float*  h0f = (float*)alloc(NLC * 4);
  __bf16* h0b = (__bf16*)alloc(NLC * 2);
  __bf16* wqb = (__bf16*)alloc((size_t)Cc * Cc * 2);
  __bf16* wkb = (__bf16*)alloc((size_t)Cc * Cc * 2);
  __bf16* wvb = (__bf16*)alloc((size_t)Cc * Cc * 2);
  __bf16* wob = (__bf16*)alloc((size_t)Cc * Cc * 2);
  __bf16* w1b = (__bf16*)alloc((size_t)MD * Cc * 2);
  __bf16* w2b[3];
  for (int t = 0; t < 3; ++t) w2b[t] = (__bf16*)alloc((size_t)MD * MD * 2);
  __bf16* w3b = (__bf16*)alloc((size_t)Cc * MD * 2);
  float*  qf  = (float*)alloc(NLC * 4);
  float*  kf  = (float*)alloc(NLC * 4);
  float*  vf  = (float*)alloc(NLC * 4);
  __bf16* qh  = (__bf16*)alloc(NLC * 2);
  __bf16* kh  = (__bf16*)alloc(NLC * 2);
  __bf16* vh  = (__bf16*)alloc(NLC * 2);
  float*  att = (float*)alloc(ATT * 4);
  __bf16* atp = (__bf16*)alloc(ATT * 2);
  float*  ofh = (float*)alloc(NLC * 4);
  __bf16* ob  = (__bf16*)alloc(NLC * 2);
  float*  opj = (float*)alloc(NLC * 4);
  float*  hfb = (float*)alloc(NLC * 4);
  __bf16* hbb = (__bf16*)alloc(NLC * 2);
  float*  y1  = (float*)alloc(NLC * 4);
  __bf16* y1p = (__bf16*)alloc((size_t)NB * (Ll + 2) * MD * 2);
  float*  y2  = (float*)alloc(NLC * 4);
  __bf16* y2b = (__bf16*)alloc(NLC * 2);
  float*  y3  = (float*)alloc(NLC * 4);

  // 1) BN + transpose
  bn_transpose<<<dim3(Ll / 32, Cc / 32, NB), dim3(32, 8), 0, stream>>>(
      x, ng, nb_, nm, nv, h0f, h0b, Cc, Ll);

  // 2) weight conversion to bf16
  int gcw = (int)(((long long)Cc * Cc + 255) / 256);
  cvt_bf16_strided<<<gcw, 256, 0, stream>>>(wq,  wqb, (long long)Cc * Cc, 1, 0);
  cvt_bf16_strided<<<gcw, 256, 0, stream>>>(wk,  wkb, (long long)Cc * Cc, 1, 0);
  cvt_bf16_strided<<<gcw, 256, 0, stream>>>(wv_, wvb, (long long)Cc * Cc, 1, 0);
  cvt_bf16_strided<<<gcw, 256, 0, stream>>>(wo,  wob, (long long)Cc * Cc, 1, 0);
  cvt_bf16_strided<<<gcw, 256, 0, stream>>>(cw1, w1b, (long long)MD * Cc, 1, 0);
  for (int t = 0; t < 3; ++t)
    cvt_bf16_strided<<<gcw, 256, 0, stream>>>(cw2, w2b[t], (long long)MD * MD, 3, t);
  cvt_bf16_strided<<<gcw, 256, 0, stream>>>(cw3, w3b, (long long)Cc * MD, 1, 0);

  int gel = (int)((NLC + 255) / 256);
  dim3 gProj((Cc + 127) / 128, (NB * Ll + 127) / 128, 1);

  // 3) Q/K/V projections: (N*L,C) x (C,C)^T
  wmma_gemm<true,  false><<<gProj, 256, 0, stream>>>(h0b, Cc, 0, wqb, Cc, 0, qf, Cc, 0, NB * Ll, Cc, Cc, 1.f);
  wmma_gemm<true,  false><<<gProj, 256, 0, stream>>>(h0b, Cc, 0, wkb, Cc, 0, kf, Cc, 0, NB * Ll, Cc, Cc, 1.f);
  wmma_gemm<true,  false><<<gProj, 256, 0, stream>>>(h0b, Cc, 0, wvb, Cc, 0, vf, Cc, 0, NB * Ll, Cc, Cc, 1.f);
  pack_heads<<<gel, 256, 0, stream>>>(qf, qh, NB, Ll, NH, DH);
  pack_heads<<<gel, 256, 0, stream>>>(kf, kh, NB, Ll, NH, DH);
  pack_heads<<<gel, 256, 0, stream>>>(vf, vh, NB, Ll, NH, DH);

  // 4) scores = Q K^T / sqrt(d_model), batched over N*H
  dim3 gSc(Ll / 128, Ll / 128, NB * NH);
  wmma_gemm<true,  false><<<gSc, 256, 0, stream>>>(
      qh, DH, (long long)Ll * DH, kh, DH, (long long)Ll * DH,
      att, Ll, (long long)Ll * Ll, Ll, Ll, DH, 0.03125f);

  // 5) softmax rows
  softmax_bf16<<<NB * NH * Ll, 256, 0, stream>>>(att, atp, Ll);

  // 6) O = softmax(att) @ V  (NN GEMM, N=64)
  dim3 gAV((DH + 127) / 128, Ll / 128, NB * NH);
  wmma_gemm<false, false><<<gAV, 256, 0, stream>>>(
      atp, Ll, (long long)Ll * Ll, vh, DH, (long long)Ll * DH,
      ofh, DH, (long long)Ll * DH, Ll, DH, Ll, 1.f);
  unpack_heads<<<gel, 256, 0, stream>>>(ofh, ob, NB, Ll, NH, DH);

  // 7) output projection + residual
  wmma_gemm<true,  false><<<gProj, 256, 0, stream>>>(ob, Cc, 0, wob, Cc, 0, opj, Cc, 0, NB * Ll, Cc, Cc, 1.f);
  residual_add<<<gel, 256, 0, stream>>>(h0f, opj, hfb, hbb, NLC);

  // 8) conv1 (1x1) + BN1 + ReLU (+ zero-padded seq buffer for conv2)
  wmma_gemm<true,  false><<<gProj, 256, 0, stream>>>(hbb, Cc, 0, w1b, Cc, 0, y1, MD, 0, NB * Ll, MD, Cc, 1.f);
  int gpad = (int)(((long long)NB * (Ll + 2) * MD + 255) / 256);
  bn_act_bf16<<<gpad, 256, 0, stream>>>(y1, bn1g, bn1b, bn1m, bn1v, cb1, y1p, NB, Ll, MD, 1, 1);

  // 9) conv2 (k=3) = 3 shifted accumulated GEMMs, batched over N
  dim3 gC2(MD / 128, Ll / 128, NB);
  wmma_gemm<true,  false><<<gC2, 256, 0, stream>>>(
      y1p + 0 * MD, MD, (long long)(Ll + 2) * MD, w2b[0], MD, 0,
      y2, MD, (long long)Ll * MD, Ll, MD, MD, 1.f);
  wmma_gemm<true,  true ><<<gC2, 256, 0, stream>>>(
      y1p + 1 * MD, MD, (long long)(Ll + 2) * MD, w2b[1], MD, 0,
      y2, MD, (long long)Ll * MD, Ll, MD, MD, 1.f);
  wmma_gemm<true,  true ><<<gC2, 256, 0, stream>>>(
      y1p + 2 * MD, MD, (long long)(Ll + 2) * MD, w2b[2], MD, 0,
      y2, MD, (long long)Ll * MD, Ll, MD, MD, 1.f);
  bn_act_bf16<<<gel, 256, 0, stream>>>(y2, bn2g, bn2b, bn2m, bn2v, cb2, y2b, NB, Ll, MD, 0, 1);

  // 10) conv3 (1x1)
  wmma_gemm<true,  false><<<gProj, 256, 0, stream>>>(y2b, MD, 0, w3b, MD, 0, y3, Cc, 0, NB * Ll, Cc, MD, 1.f);

  // 11) BN3 + residual + instance-norm + ReLU + maxpool2
  final_fuse<<<NB * Cc, 256, 0, stream>>>(y3, hfb, bn3g, bn3b, bn3m, bn3v, cb3,
                                          (float*)d_out, Cc, Ll);
}
